// plgMultiHeadAttention_81123342287386
// MI455X (gfx1250) — matrix operand order repaired and checked by reading, not compiled, not run on hardware
//
#include <hip/hip_runtime.h>
#include <hip/hip_bf16.h>
#include <math.h>

typedef __bf16 bf16_t;
typedef __attribute__((ext_vector_type(8)))  __bf16 bf16x8;
typedef __attribute__((ext_vector_type(16))) __bf16 bf16x16;
typedef __attribute__((ext_vector_type(8)))  float  f32x8;
typedef __attribute__((ext_vector_type(4)))  unsigned int u32x4;
typedef __attribute__((ext_vector_type(8)))  int i32x8;
typedef __attribute__((ext_vector_type(4)))  int i32x4;

#define DEV static __device__ __forceinline__

static constexpr int  Bb = 2, Ss = 2048, Dd = 2048, Hh = 16, DH = 128;
static constexpr float SCALE = 0.08838834764831845f;   // 1/sqrt(128)

#if defined(__AMDGCN__) && __has_builtin(__builtin_amdgcn_tensor_load_to_lds) && \
    __has_builtin(__builtin_amdgcn_s_wait_tensorcnt)
#define HAVE_TDM 1
#else
#define HAVE_TDM 0
#endif

DEV f32x8 fzero() { f32x8 z;
#pragma unroll
  for (int i = 0; i < 8; ++i) z[i] = 0.f;
  return z; }

// A-operand fragment (16xK32, 16-bit): lane m=l&15; elems 0..7 -> K=kb..kb+7,
// elems 8..15 -> K=kb+16..kb+23, kb = 8*(l>=16).   (ISA 7.12.2)
DEV bf16x16 frag_a(const bf16_t* t, int ld) {
  const int l = threadIdx.x & 31;
  const bf16_t* p = t + (l & 15) * ld + ((l >> 4) << 3);
  bf16x8 lo = *(const bf16x8*)(p);
  bf16x8 hi = *(const bf16x8*)(p + 16);
  bf16x16 r;
#pragma unroll
  for (int i = 0; i < 8; ++i) { r[i] = lo[i]; r[i + 8] = hi[i]; }
  return r;
}

// B-operand fragment (K32x16, 16-bit) read from a row-major [N][K] tile
// (i.e. already-transposed B): lane n=l&15; elems 0..7 -> K=kb..kb+7,
// elems 8..15 -> K=kb+16..kb+23, kb = 8*(l>=16).
DEV bf16x16 frag_b(const bf16_t* t, int ld) {
  const int l = threadIdx.x & 31;
  const bf16_t* p = t + (l & 15) * ld + ((l >> 4) << 3);
  bf16x8 lo = *(const bf16x8*)(p);
  bf16x8 hi = *(const bf16x8*)(p + 16);
  bf16x16 r;
#pragma unroll
  for (int i = 0; i < 8; ++i) { r[i] = lo[i]; r[i + 8] = hi[i]; }
  return r;
}

DEV f32x8 wmma_bf(bf16x16 a, bf16x16 b, f32x8 c) {
  return __builtin_amdgcn_wmma_f32_16x16x32_bf16(false, a, false, b, (short)0, c,
                                                 false, false);
}

#if HAVE_TDM
// TDM: DMA a [64 rows x 32 cols] bf16 tile (row stride = ld elements in global)
// into LDS at lds_byte, with HW padding producing a 40-element LDS row stride
// (pad_interval = 16 DWORDs, pad_amount = 4 DWORDs).  ISA 7.2/8.x.
DEV void tdm_w_tile(unsigned lds_byte, const bf16_t* gsrc, int ld) {
  unsigned long ga = (unsigned long)gsrc;
  u32x4 g0;
  g0[0] = 1u;                                     // count=1, user mode
  g0[1] = lds_byte;                               // lds_addr (generic low32 == LDS byte)
  g0[2] = (unsigned)ga;                           // global_addr[31:0]
  g0[3] = (unsigned)((ga >> 32) & 0x1FFFFFFu) | (2u << 30);  // addr[56:32] | type=2
  unsigned td0 = (unsigned)ld;                    // tensor_dim0 = full row length
  unsigned long st0 = (unsigned long)ld;          // tensor_dim0_stride
  i32x8 g1;
  g1[0] = (int)((1u << 16) | (1u << 20) | (3u << 22) | (3u << 25));
  //              data=2B    pad_en       intv=16dw     amt=4dw
  g1[1] = (int)((td0 & 0xFFFFu) << 16);           // tensor_dim0[15:0] at bits 63:48
  g1[2] = (int)(((td0 >> 16) & 0xFFFFu) | (64u << 16)); // dim0 hi | tensor_dim1 lo
  g1[3] = (int)(32u << 16);                       // tensor_dim1 hi=0 | tile_dim0=32
  g1[4] = (int)64u;                               // tile_dim1=64, tile_dim2=0
  g1[5] = (int)(unsigned)st0;                     // dim0_stride[31:0]
  g1[6] = (int)((unsigned)(st0 >> 32) & 0xFFFFu); // dim0_stride[47:32] | dim1_stride lo=0
  g1[7] = 0;
  i32x4 gz; gz[0] = gz[1] = gz[2] = gz[3] = 0;
#if __clang_major__ >= 23
  i32x8 gz8;
#pragma unroll
  for (int i = 0; i < 8; ++i) gz8[i] = 0;
  __builtin_amdgcn_tensor_load_to_lds(g0, g1, gz, gz, gz8, 0);
#else
  __builtin_amdgcn_tensor_load_to_lds(g0, g1, gz, gz, 0);
#endif
}
#endif

// ---------------------------------------------------------------- conversions
__global__ void k_cvt4(const float4* __restrict__ in, bf16_t* __restrict__ out,
                       int n4) {
  int i = blockIdx.x * blockDim.x + threadIdx.x;
  if (i < n4) {
    float4 v = in[i];
    bf16_t* o = out + (long)i * 4;
    o[0] = (bf16_t)v.x; o[1] = (bf16_t)v.y; o[2] = (bf16_t)v.z; o[3] = (bf16_t)v.w;
  }
}

// ------------------------------------------------- generic C = X @ W^T + bias
// X:[M,K] bf16, W:[N,K] bf16 (row-major == pre-transposed B), C f32 or bf16.
// block = 256 thr (8 waves): (128*MW)x64 C tile; wave w -> 16*MW rows, 4 N-tiles.
// W tile double-buffered in LDS; TDM (wave 0) DMAs tile k+32 while computing k.
template <int MW>
__global__ void __launch_bounds__(256)
k_gemm_xwt(const bf16_t* __restrict__ X, const bf16_t* __restrict__ W,
           const float* __restrict__ bias, float* __restrict__ Cf,
           bf16_t* __restrict__ Cb, int M, int N, int K,
           long sx, long sw, long sc) {
  __shared__ bf16_t Ws[2][64 * 40];
  const int wave = threadIdx.x >> 5, lane = threadIdx.x & 31;
  const bf16_t* Xb = X + blockIdx.z * sx +
                     (long)(blockIdx.y * 128 * MW + wave * 16 * MW) * K;
  const bf16_t* Wb = W + blockIdx.z * sw + (long)(blockIdx.x * 64) * K;
  f32x8 acc[MW][4];
#pragma unroll
  for (int mi = 0; mi < MW; ++mi)
#pragma unroll
    for (int j = 0; j < 4; ++j) acc[mi][j] = fzero();

#if HAVE_TDM
  if (wave == 0) tdm_w_tile((unsigned)(unsigned long)&Ws[0][0], Wb, K);
#endif
  const int srow = threadIdx.x >> 2, sch = (threadIdx.x & 3) << 3;
  (void)srow; (void)sch;
  for (int k0 = 0; k0 < K; k0 += 32) {
    const int buf = (k0 >> 5) & 1;
#if HAVE_TDM
    if (wave == 0) __builtin_amdgcn_s_wait_tensorcnt(0);  // current buf landed
    __syncthreads();              // publish LDS; all done reading buf^1
    if (wave == 0 && k0 + 32 < K)
      tdm_w_tile((unsigned)(unsigned long)&Ws[buf ^ 1][0], Wb + k0 + 32, K);
#else
    if (k0) __syncthreads();
    *(bf16x8*)(&Ws[buf][srow * 40 + sch]) =
        *(const bf16x8*)(Wb + (long)srow * K + k0 + sch);
    __syncthreads();
#endif
    if (k0 + 32 < K) __builtin_prefetch(Xb + k0 + 32, 0, 1);
#pragma unroll
    for (int mi = 0; mi < MW; ++mi) {
      bf16x16 af = frag_a(Xb + (long)(mi * 16) * K + k0, K);
#pragma unroll
      for (int j = 0; j < 4; ++j)
        acc[mi][j] = wmma_bf(af, frag_b(&Ws[buf][j * 16 * 40], 40), acc[mi][j]);
    }
  }
  const int hi = lane >> 4, col = lane & 15;
#pragma unroll
  for (int mi = 0; mi < MW; ++mi)
#pragma unroll
    for (int j = 0; j < 4; ++j) {
      int n = blockIdx.x * 64 + j * 16 + col;
      float bv = bias ? bias[n] : 0.f;
#pragma unroll
      for (int r = 0; r < 8; ++r) {
        long m = blockIdx.y * 128 * MW + wave * 16 * MW + mi * 16 + r + hi * 8;
        long idx = blockIdx.z * sc + m * (long)N + n;
        float v = acc[mi][j][r] + bv;
        if (Cf) Cf[idx] = v; else Cb[idx] = (bf16_t)v;
      }
    }
}

// -------------------------------------------- RoPE + head split / transposes
// mode 0: rope, write SD=[B,H,S,DH] and DS=[B,H,DH,S]
// mode 1: rope, write SD only;  mode 2: no rope, write DS only
__global__ void k_rope_scatter(const bf16_t* __restrict__ src,
                               bf16_t* __restrict__ dstSD,
                               bf16_t* __restrict__ dstDS, int mode) {
  long gid = blockIdx.x * (long)blockDim.x + threadIdx.x;
  int  i   = gid & 63;             // pair index (half = 64)
  long rest = gid >> 6;
  int  h   = rest & 15;
  long bs  = rest >> 4;            // b*S + s
  int  s   = (int)(bs & (Ss - 1));
  const bf16_t* p = src + (bs << 11) + h * DH + i * 2;
  float x1 = (float)p[0], x2 = (float)p[1];
  float o1 = x1, o2 = x2;
  if (mode != 2) {
    float freq = __powf(10000.f, -(2.f * i) / (float)DH);
    float ang = (float)s * freq;
    float c = cosf(ang), sn = sinf(ang);
    o1 = x1 * c - x2 * sn;
    o2 = x2 * c + x1 * sn;
  }
  long head = (bs >> 11) * Hh + h;
  if (mode != 2) {
    bf16_t* q = dstSD + head * (long)Ss * DH + (long)s * DH + i * 2;
    q[0] = (bf16_t)o1; q[1] = (bf16_t)o2;
  }
  if (mode != 1) {
    bf16_t* q = dstDS + head * (long)Ss * DH + (long)(i * 2) * Ss + s;
    q[0]  = (bf16_t)o1;
    q[Ss] = (bf16_t)o2;
  }
}

// ------------------------------- per-(b,h) metric stack: LN + SwiGLU x16 + plga
DEV void ln_rows(float* Af, const float* g, const float* b) {
  const int t = threadIdx.x;
  const int row = t >> 1, half = t & 1;
  float* p = Af + row * 128 + half * 64;
  float s = 0.f;
  for (int i = 0; i < 64; ++i) s += p[i];
  s += __shfl_xor(s, 1, 32);
  float mu = s * (1.f / 128.f);
  float v = 0.f;
  for (int i = 0; i < 64; ++i) { float d = p[i] - mu; v += d * d; }
  v += __shfl_xor(v, 1, 32);
  float inv = rsqrtf(v * (1.f / 128.f) + 1e-6f);
  for (int i = 0; i < 64; ++i) {
    int c = half * 64 + i;
    p[i] = (p[i] - mu) * inv * g[c] + b[c];
  }
}

__global__ void __launch_bounds__(256)
k_metric(float* __restrict__ Abuf, const float* __restrict__ g1,
         const float* __restrict__ bb1, const bf16_t* __restrict__ w1,
         const bf16_t* __restrict__ w2, const bf16_t* __restrict__ w3,
         const float* __restrict__ rb1, const float* __restrict__ rb2,
         const float* __restrict__ rb3, const float* __restrict__ rg,
         const float* __restrict__ rb, const bf16_t* __restrict__ pW,
         const float* __restrict__ pB, const float* __restrict__ pPw,
         bf16_t* __restrict__ G) {
  extern __shared__ unsigned char smem[];
  float*  Af  = (float*)smem;                               // 128*128 f32
  bf16_t* Abf = (bf16_t*)(smem + 128 * 128 * 4);            // 128*136 bf16
  bf16_t* Hb  = (bf16_t*)(smem + 128 * 128 * 4 + 128 * 136 * 2); // 128*264 bf16

  const int t = threadIdx.x, wave = t >> 5, lane = t & 31;
  const int hi = lane >> 4, col = lane & 15;
  const int bh = blockIdx.x, h = bh & 15;
  float* Ag = Abuf + (long)bh * 16384;

  for (int i = t; i < 16384; i += 256) Af[i] = Ag[i];
  __syncthreads();
  ln_rows(Af, g1, bb1);
  __syncthreads();

  float res[64];
  for (int l = 0; l < 8; ++l) {
#pragma unroll 8
    for (int i = 0; i < 64; ++i) res[i] = Af[t * 64 + i];
    for (int n = 0; n < 2; ++n) {
#pragma unroll 8
      for (int i = 0; i < 64; ++i) {
        int idx = t * 64 + i;
        Abf[(idx >> 7) * 136 + (idx & 127)] = (bf16_t)Af[idx];
      }
      __syncthreads();
      const int ln = l * 2 + n;
      const bf16_t* W1 = w1 + (long)ln * 256 * 128;
      const bf16_t* W2 = w2 + (long)ln * 256 * 128;
      const float*  B1 = rb1 + ln * 256;
      const float*  B2 = rb2 + ln * 256;
      const bf16_t* Arow = &Abf[wave * 16 * 136];
      for (int j = 0; j < 16; ++j) {          // 256-wide hidden, 16 cols/tile
        f32x8 a1 = fzero(), a2 = fzero();
#pragma unroll
        for (int k = 0; k < 4; ++k) {
          bf16x16 af = frag_a(Arow + k * 32, 136);
          a1 = wmma_bf(af, frag_b(W1 + (long)(j * 16) * 128 + k * 32, 128), a1);
          a2 = wmma_bf(af, frag_b(W2 + (long)(j * 16) * 128 + k * 32, 128), a2);
        }
        int o = j * 16 + col;
        float c1 = B1[o], c2 = B2[o];
#pragma unroll
        for (int r = 0; r < 8; ++r) {
          float x1 = a1[r] + c1, x2 = a2[r] + c2;
          float sl = x1 / (1.f + __expf(-x1));
          Hb[(wave * 16 + r + hi * 8) * 264 + o] = (bf16_t)(sl * x2);
        }
      }
      __syncthreads();
      const bf16_t* W3 = w3 + (long)ln * 128 * 256;
      const float*  B3 = rb3 + ln * 128;
      const bf16_t* Hrow = &Hb[wave * 16 * 264];
      for (int j = 0; j < 8; ++j) {
        f32x8 a3 = fzero();
#pragma unroll
        for (int k = 0; k < 8; ++k)
          a3 = wmma_bf(frag_a(Hrow + k * 32, 264),
                       frag_b(W3 + (long)(j * 16) * 256 + k * 32, 256), a3);
        int o = j * 16 + col;
        float c3 = B3[o];
#pragma unroll
        for (int r = 0; r < 8; ++r)
          Af[(wave * 16 + r + hi * 8) * 128 + o] = a3[r] + c3;
      }
      __syncthreads();
    }
#pragma unroll 8
    for (int i = 0; i < 64; ++i) Af[t * 64 + i] += res[i];
    __syncthreads();
    ln_rows(Af, rg + l * 128, rb + l * 128);
    __syncthreads();
  }

  // plga: A_LM = leaky(W[h] @ A + b); need A^T as B-operand -> stage in Hb space
  bf16_t* AbT = Hb;  // [f][e], stride 136
#pragma unroll 8
  for (int i = 0; i < 64; ++i) {
    int idx = t * 64 + i;
    AbT[(idx & 127) * 136 + (idx >> 7)] = (bf16_t)Af[idx];
  }
  __syncthreads();
  const bf16_t* Wp = pW + (long)h * 16384 + (long)(wave * 16) * 128;
  float pw = pPw[h];
  for (int j = 0; j < 8; ++j) {
    f32x8 a = fzero();
#pragma unroll
    for (int k = 0; k < 4; ++k)
      a = wmma_bf(frag_a(Wp + k * 32, 128),
                  frag_b(&AbT[(j * 16) * 136 + k * 32], 136), a);
    int f = j * 16 + col;
    float bb = pB[h * 128 + f];
#pragma unroll
    for (int r = 0; r < 8; ++r) {
      float x = a[r] + bb;
      x = x >= 0.f ? x : 0.2f * x;
      float gv = (x >= 0.f ? 1.f : -1.f) * __powf(fabsf(x) + 1e-6f, pw);
      G[(long)bh * 16384 + (long)(wave * 16 + r + hi * 8) * 128 + f] = (bf16_t)gv;
    }
  }
}

// ------------------------------------ causal flash attention with metric keys
// scores = q @ P^T (P = k@G^T precomputed), online softmax, O = att @ v
__global__ void __launch_bounds__(256)
k_attn(const bf16_t* __restrict__ Qh, const bf16_t* __restrict__ P,
       const bf16_t* __restrict__ VT, bf16_t* __restrict__ Hn) {
  __shared__ bf16_t sS[8][16 * 40];
  const int bh = blockIdx.x;
  const int wave = threadIdx.x >> 5, lane = threadIdx.x & 31;
  const int hi = lane >> 4, col = lane & 15;
  const long hoff = (long)bh * Ss * DH;
  const int s0 = blockIdx.y * 128 + wave * 16;

  bf16x16 qf[4];
  const bf16_t* Qrow = Qh + hoff + (long)s0 * DH;
#pragma unroll
  for (int k = 0; k < 4; ++k) qf[k] = frag_a(Qrow + k * 32, DH);

  f32x8 O[8];
#pragma unroll
  for (int j = 0; j < 8; ++j) O[j] = fzero();
  float m[8], lsum[8];
#pragma unroll
  for (int r = 0; r < 8; ++r) { m[r] = -1e30f; lsum[r] = 0.f; }

  const int ktend = blockIdx.y * 128 + 128;   // block-uniform causal bound
  for (int kt = 0; kt < ktend; kt += 32) {
    f32x8 sc[2];
#pragma unroll
    for (int c2 = 0; c2 < 2; ++c2) {
      sc[c2] = fzero();
      const bf16_t* Pt = P + hoff + (long)(kt + c2 * 16) * DH;
#pragma unroll
      for (int k = 0; k < 4; ++k)
        sc[c2] = wmma_bf(qf[k], frag_b(Pt + k * 32, DH), sc[c2]);
    }
    float x[2][8], rmax[8];
#pragma unroll
    for (int c2 = 0; c2 < 2; ++c2)
#pragma unroll
      for (int r = 0; r < 8; ++r) {
        int tc = kt + c2 * 16 + col, sr = s0 + r + hi * 8;
        x[c2][r] = sc[c2][r] * SCALE + (tc > sr ? -1e9f : 0.f);
      }
#pragma unroll
    for (int r = 0; r < 8; ++r) rmax[r] = fmaxf(x[0][r], x[1][r]);
#pragma unroll
    for (int dd = 1; dd < 16; dd <<= 1)
#pragma unroll
      for (int r = 0; r < 8; ++r)
        rmax[r] = fmaxf(rmax[r], __shfl_xor(rmax[r], dd, 32));
    float fac[8], psum[8];
#pragma unroll
    for (int r = 0; r < 8; ++r) {
      float nm = fmaxf(m[r], rmax[r]);
      fac[r] = __expf(m[r] - nm);
      m[r] = nm;
      psum[r] = 0.f;
    }
#pragma unroll
    for (int c2 = 0; c2 < 2; ++c2)
#pragma unroll
      for (int r = 0; r < 8; ++r) {
        float p = __expf(x[c2][r] - m[r]);
        psum[r] += p;
        sS[wave][(r + hi * 8) * 40 + c2 * 16 + col] = (bf16_t)p;
      }
#pragma unroll
    for (int dd = 1; dd < 16; dd <<= 1)
#pragma unroll
      for (int r = 0; r < 8; ++r) psum[r] += __shfl_xor(psum[r], dd, 32);
#pragma unroll
    for (int r = 0; r < 8; ++r) lsum[r] = lsum[r] * fac[r] + psum[r];
#pragma unroll
    for (int j = 0; j < 8; ++j)
#pragma unroll
      for (int r = 0; r < 8; ++r) O[j][r] *= fac[r];
    __syncthreads();                       // transpose staging visibility
    bf16x16 pa = frag_a(&sS[wave][0], 40); // p-tile as A-operand (16x32)
#pragma unroll
    for (int j = 0; j < 8; ++j)
      O[j] = wmma_bf(pa, frag_b(VT + (long)bh * DH * Ss + (long)(j * 16) * Ss + kt, Ss),
                     O[j]);
    __syncthreads();
  }

  const int b = bh >> 4, h = bh & 15;
#pragma unroll
  for (int j = 0; j < 8; ++j)
#pragma unroll
    for (int r = 0; r < 8; ++r) {
      int sr = s0 + r + hi * 8;
      float o = O[j][r] / lsum[r];
      Hn[((long)b * Ss + sr) * Dd + h * DH + j * 16 + col] = (bf16_t)o;
    }
}

// ---------------------------------------------------------------------- host
extern "C" void kernel_launch(void* const* d_in, const int* in_sizes, int n_in,
                              void* d_out, int out_size, void* d_ws, size_t ws_size,
                              hipStream_t stream) {
  (void)in_sizes; (void)n_in; (void)out_size; (void)ws_size;
  const long EL = (long)Bb * Ss * Dd;          // 8,388,608
  unsigned char* ws = (unsigned char*)d_ws;
  size_t off = 0;
  auto alloc = [&](size_t bytes) { size_t r = off; off = (off + bytes + 255) & ~(size_t)255; return r; };

  bf16_t* qbf = (bf16_t*)(ws + alloc(EL * 2));
  bf16_t* kbf = (bf16_t*)(ws + alloc(EL * 2));
  bf16_t* vbf = (bf16_t*)(ws + alloc(EL * 2));
  bf16_t* wqb = (bf16_t*)(ws + alloc((size_t)Dd * Dd * 2));
  bf16_t* wkb = (bf16_t*)(ws + alloc((size_t)Dd * Dd * 2));
  bf16_t* wvb = (bf16_t*)(ws + alloc((size_t)Dd * Dd * 2));
  bf16_t* wdb = (bf16_t*)(ws + alloc((size_t)Dd * Dd * 2));
  bf16_t* w1b = (bf16_t*)(ws + alloc((size_t)8 * 2 * 256 * 128 * 2));
  bf16_t* w2b = (bf16_t*)(ws + alloc((size_t)8 * 2 * 256 * 128 * 2));
  bf16_t* w3b = (bf16_t*)(ws + alloc((size_t)8 * 2 * 128 * 256 * 2));
  bf16_t* pWb = (bf16_t*)(ws + alloc((size_t)Hh * DH * DH * 2));
  bf16_t* qp  = (bf16_t*)(ws + alloc(EL * 2));
  bf16_t* kp  = (bf16_t*)(ws + alloc(EL * 2));
  bf16_t* vp  = (bf16_t*)(ws + alloc(EL * 2));
  float*  Abuf= (float*)(ws + alloc((size_t)Bb * Hh * DH * DH * 4));
  bf16_t* Gb  = (bf16_t*)(ws + alloc((size_t)Bb * Hh * DH * DH * 2));
  // aliases (buffers whose producers have been fully consumed):
  bf16_t* Qh = qbf;   // [B,H,S,DH]
  bf16_t* QT = kbf;   // [B,H,DH,S]
  bf16_t* Kh = vbf;   // [B,H,S,DH]
  bf16_t* VT = qp;    // [B,H,DH,S]
  bf16_t* Pm = kp;    // [B,H,S,DH]  (k @ G^T)
  bf16_t* Hn = vp;    // [B,S,D]

  auto cvt = [&](const void* src, bf16_t* dst, long n) {
    k_cvt4<<<dim3((unsigned)((n / 4 + 255) / 256)), 256, 0, stream>>>(
        (const float4*)src, dst, (int)(n / 4));
  };
  cvt(d_in[0], qbf, EL);  cvt(d_in[1], kbf, EL);  cvt(d_in[2], vbf, EL);
  cvt(d_in[4], wqb, (long)Dd * Dd);  cvt(d_in[6], wkb, (long)Dd * Dd);
  cvt(d_in[8], wvb, (long)Dd * Dd);  cvt(d_in[10], wdb, (long)Dd * Dd);
  cvt(d_in[14], w1b, 8L * 2 * 256 * 128);
  cvt(d_in[16], w2b, 8L * 2 * 256 * 128);
  cvt(d_in[18], w3b, 8L * 2 * 128 * 256);
  cvt(d_in[22], pWb, (long)Hh * DH * DH);

  // QKV projections: C = X @ W^T + b, bf16 out (256-row block tiles)
  k_gemm_xwt<2><<<dim3(32, 16, 1), 256, 0, stream>>>(qbf, wqb, (const float*)d_in[5],
      nullptr, qp, Bb * Ss, Dd, Dd, 0, 0, 0);
  k_gemm_xwt<2><<<dim3(32, 16, 1), 256, 0, stream>>>(kbf, wkb, (const float*)d_in[7],
      nullptr, kp, Bb * Ss, Dd, Dd, 0, 0, 0);
  k_gemm_xwt<2><<<dim3(32, 16, 1), 256, 0, stream>>>(vbf, wvb, (const float*)d_in[9],
      nullptr, vp, Bb * Ss, Dd, Dd, 0, 0, 0);

  // RoPE + head split / transposes
  const unsigned nsc = (unsigned)((long)Bb * Ss * Hh * 64 / 256);
  k_rope_scatter<<<nsc, 256, 0, stream>>>(qp, Qh, QT, 0);
  k_rope_scatter<<<nsc, 256, 0, stream>>>(kp, Kh, nullptr, 1);
  k_rope_scatter<<<nsc, 256, 0, stream>>>(vp, nullptr, VT, 2);

  // A = Q^T Q per head: batched 128x128x2048, f32 out (1 frag/wave variant)
  k_gemm_xwt<1><<<dim3(2, 1, Bb * Hh), 256, 0, stream>>>(QT, QT, nullptr, Abuf,
      nullptr, DH, DH, Ss, (long)DH * Ss, (long)DH * Ss, (long)DH * DH);

  // metric stack (LN + SwiGLU residual blocks + plga + power) -> G bf16
  const size_t shmem = 128 * 128 * 4 + 128 * 136 * 2 + 128 * 264 * 2; // 167936B
  hipFuncSetAttribute(reinterpret_cast<const void*>(k_metric),
                      hipFuncAttributeMaxDynamicSharedMemorySize, (int)shmem);
  k_metric<<<Bb * Hh, 256, shmem, stream>>>(Abuf,
      (const float*)d_in[12], (const float*)d_in[13], w1b, w2b, w3b,
      (const float*)d_in[15], (const float*)d_in[17], (const float*)d_in[19],
      (const float*)d_in[20], (const float*)d_in[21], pWb,
      (const float*)d_in[23], (const float*)d_in[24], Gb);

  // P = K @ G^T per head: batched 2048x128x128, bf16 out
  k_gemm_xwt<2><<<dim3(2, 8, Bb * Hh), 256, 0, stream>>>(Kh, Gb, nullptr, nullptr,
      Pm, Ss, DH, DH, (long)Ss * DH, (long)DH * DH, (long)Ss * DH);

  // causal flash attention with metric keys
  k_attn<<<dim3(Bb * Hh, Ss / 128), 256, 0, stream>>>(Qh, Pm, VT, Hn);

  // output dense: f32 -> d_out
  k_gemm_xwt<2><<<dim3(32, 16, 1), 256, 0, stream>>>(Hn, wdb, (const float*)d_in[11],
      (float*)d_out, nullptr, Bb * Ss, Dd, Dd, 0, 0, 0);
}
